// EdgeMLP_36971078484654
// MI455X (gfx1250) — matrix operand order repaired and checked by reading, compiled, not run on hardware
//
#include <hip/hip_runtime.h>

typedef _Float16 v16h  __attribute__((ext_vector_type(16)));
typedef _Float16 half8 __attribute__((ext_vector_type(8)));
typedef _Float16 half4 __attribute__((ext_vector_type(4)));
typedef float    v8f   __attribute__((ext_vector_type(8)));

#define DFEAT     64
#define HID       128
#define EPW       16      // edges per wave (M tile)
#define EPB       128     // edges per block
#define THREADS   256     // 8 waves
#define XS_STRIDE 264     // 256 + 8 halves pad (bank-conflict avoidance)
#define HS_STRIDE 136     // 128 + 8 halves pad

// f16 fragment-ordered weights in d_ws (counts in halves)
#define W0F_CNT (8 * 8 * 512)          // 32768
#define W1F_CNT (4 * 8 * 512)          // 16384
#define W2F_CNT (4 * 512)              // 2048
#define W1F_OFF (W0F_CNT)              // 32768
#define W2F_OFF (W0F_CNT + W1F_CNT)    // 49152
#define WS_HALVES (W0F_CNT + W1F_CNT + W2F_CNT)  // 51200

// ---------------------------------------------------------------------------
// Prep: f32 weights -> f16, pre-swizzled into WMMA B-fragment layout.
// B 32x16 (KxN), contiguous k order used here (matches A): lane n (0..15) holds column N=n, K=0..15 packed
// 2-per-VGPR; lane n+16 holds K=16..31.  Fragment = 32 lanes * 16 halves.
// W2 (128x1) is zero-padded to 16 columns (only N==0 live).
// ---------------------------------------------------------------------------
__global__ void prep_w_36971078484654(const float* __restrict__ W0,
                                      const float* __restrict__ W1,
                                      const float* __restrict__ W2,
                                      _Float16* __restrict__ wsh) {
  int t = blockIdx.x * blockDim.x + threadIdx.x;
  if (t >= WS_HALVES) return;
  float val;
  if (t < W1F_OFF) {                       // W0: [256,128], 8 kt x 8 nt frags
    int f = t >> 9, r = t & 511;
    int lane = r >> 4, h = r & 15;
    int kt = f >> 3, nt = f & 7;
    int K = kt * 32 + ((lane >> 4) << 4) + h;
    int N = nt * 16 + (lane & 15);
    val = W0[K * HID + N];
  } else if (t < W2F_OFF) {                // W1: [128,128], 4 kt x 8 nt frags
    int u = t - W1F_OFF;
    int f = u >> 9, r = u & 511;
    int lane = r >> 4, h = r & 15;
    int kt = f >> 3, nt = f & 7;
    int K = kt * 32 + ((lane >> 4) << 4) + h;
    int N = nt * 16 + (lane & 15);
    val = W1[K * HID + N];
  } else {                                 // W2: [128,1] zero-padded, 4 kt frags
    int u = t - W2F_OFF;
    int f = u >> 9, r = u & 511;
    int lane = r >> 4, h = r & 15;
    int K = f * 32 + ((lane >> 4) << 4) + h;
    val = ((lane & 15) == 0) ? W2[K] : 0.0f;
  }
  wsh[t] = (_Float16)val;
}

// A 16x32 (MxK), contiguous k order used here (matches B): lane L (0..15) row M=L holds K=0..15;
// lane L+16 holds K=16..31.  Two 16B LDS chunks per lane.
__device__ __forceinline__ v16h load_a_frag(const _Float16* row, int kb, int laneHi) {
  half8 lo = *(const half8*)(row + kb + laneHi * 16);
  half8 hi = *(const half8*)(row + kb + 8 + laneHi * 16);
  v16h a;
#pragma unroll
  for (int j = 0; j < 8; ++j) { a[j] = lo[j]; a[j + 8] = hi[j]; }
  return a;
}

__device__ __forceinline__ v8f wmma_f16(v16h a, v16h b, v8f c) {
  return __builtin_amdgcn_wmma_f32_16x16x32_f16(false, a, false, b,
                                                (short)0, c, false, false);
}

__launch_bounds__(THREADS, 1)
__global__ void edge_mlp_36971078484654(const float* __restrict__ h1,
                                        const float* __restrict__ h2,
                                        const int*   __restrict__ srcI,
                                        const int*   __restrict__ dstI,
                                        const float* __restrict__ b0,
                                        const float* __restrict__ b1,
                                        const float* __restrict__ b2,
                                        const _Float16* __restrict__ wsh,
                                        float* __restrict__ out, int nEdges) {
  __shared__ __align__(32) _Float16 Xs[EPB * XS_STRIDE];   //  67,584 B (reused as Hs2)
  __shared__ __align__(32) _Float16 Hs[EPB * HS_STRIDE];   //  34,816 B
  __shared__ __align__(32) _Float16 Wlds[WS_HALVES];       // 102,400 B

  const int tid    = threadIdx.x;
  const int lane   = tid & 31;
  const int wave   = tid >> 5;
  const int laneLo = lane & 15;
  const int laneHi = lane >> 4;
  const int e0     = blockIdx.x * EPB;

  // ---- Stage f16 fragment-ordered weights into LDS (coalesced b128 copies) -
#pragma unroll
  for (int i = 0; i < WS_HALVES / (THREADS * 8); ++i) {    // 25 iters
    int idx = (i * THREADS + tid) * 8;
    *(half8*)(&Wlds[idx]) = *(const half8*)(&wsh[idx]);
  }

  // ---- Stage gathered edge features into LDS as f16 ------------------------
  // 4 segs (h1[src], h1[dst], h2[src], h2[dst]) x 128 edges x 16 float4.
  // 16 consecutive lanes read one 256B contiguous node row -> coalesced.
#pragma unroll 8
  for (int i = 0; i < 32; ++i) {
    int g    = i * THREADS + tid;            // 0..8191
    int lp   = g & 15;
    int edge = (g >> 4) & 127;
    int seg  = g >> 11;
    int eg   = e0 + edge;
    if (eg >= nEdges) eg = nEdges - 1;       // clamp (padding rows are garbage)
    int node = (seg & 1) ? dstI[eg] : srcI[eg];
    const float4* p = (const float4*)(((seg < 2) ? h1 : h2) + (size_t)node * DFEAT) + lp;
    float4 v = *p;
    half4 hv;
    hv[0] = (_Float16)v.x; hv[1] = (_Float16)v.y;
    hv[2] = (_Float16)v.z; hv[3] = (_Float16)v.w;
    *(half4*)(&Xs[edge * XS_STRIDE + seg * 64 + lp * 4]) = hv;
  }
  __syncthreads();

  const int eB = wave * EPW;                          // wave's edge strip
  const _Float16* xrow = &Xs[(eB + laneLo) * XS_STRIDE];

  // ---- Layer 0: [16,256] x [256,128] ---------------------------------------
  v16h a0[8];
#pragma unroll
  for (int kt = 0; kt < 8; ++kt) a0[kt] = load_a_frag(xrow, kt * 32, laneHi);

  // All Xs reads are done (a0 in registers); after this barrier Xs may be
  // reused as Hs2 by any wave without racing another wave's Xs reads.
  __syncthreads();

  const v16h* w0v = (const v16h*)(&Wlds[0]);
#pragma unroll
  for (int nt = 0; nt < 8; ++nt) {
    v8f acc = {};
#pragma unroll
    for (int kt = 0; kt < 8; ++kt) {
      v16h b = w0v[(kt * 8 + nt) * 32 + lane];        // 32B/lane from LDS
      acc = wmma_f16(a0[kt], b, acc);
    }
    float bias = b0[nt * 16 + laneLo];
#pragma unroll
    for (int v = 0; v < 8; ++v) {                     // C: M=v+8*laneHi, N=laneLo
      float x = acc[v] + bias;
      x = x > 0.0f ? x : 0.0f;
      Hs[(eB + v + laneHi * 8) * HS_STRIDE + nt * 16 + laneLo] = (_Float16)x;
    }
  }

  // ---- Layer 1: [16,128] x [128,128] ---------------------------------------
  const _Float16* hrow = &Hs[(eB + laneLo) * HS_STRIDE];
  v16h a1[4];
#pragma unroll
  for (int kt = 0; kt < 4; ++kt) a1[kt] = load_a_frag(hrow, kt * 32, laneHi);

  _Float16* Hs2 = Xs;                                 // reuse Xs region
  const v16h* w1v = (const v16h*)(&Wlds[W1F_OFF]);
#pragma unroll
  for (int nt = 0; nt < 8; ++nt) {
    v8f acc = {};
#pragma unroll
    for (int kt = 0; kt < 4; ++kt) {
      v16h b = w1v[(kt * 8 + nt) * 32 + lane];
      acc = wmma_f16(a1[kt], b, acc);
    }
    float bias = b1[nt * 16 + laneLo];
#pragma unroll
    for (int v = 0; v < 8; ++v) {
      float x = acc[v] + bias;
      x = x > 0.0f ? x : 0.0f;
      Hs2[(eB + v + laneHi * 8) * HS_STRIDE + nt * 16 + laneLo] = (_Float16)x;
    }
  }

  // ---- Layer 2: [16,128] x [128,1] (N padded to 16, only col 0 live) -------
  const _Float16* h2row = &Hs2[(eB + laneLo) * HS_STRIDE];
  v16h a2[4];
#pragma unroll
  for (int kt = 0; kt < 4; ++kt) a2[kt] = load_a_frag(h2row, kt * 32, laneHi);

  const v16h* w2v = (const v16h*)(&Wlds[W2F_OFF]);
  v8f acc = {};
#pragma unroll
  for (int kt = 0; kt < 4; ++kt) {
    v16h b = w2v[kt * 32 + lane];
    acc = wmma_f16(a2[kt], b, acc);
  }
  float bias2 = b2[0];
  if (laneLo == 0) {                                  // column N=0: lanes 0 & 16
#pragma unroll
    for (int v = 0; v < 8; ++v) {
      int eg = e0 + eB + v + laneHi * 8;
      if (eg < nEdges) out[eg] = acc[v] + bias2;
    }
  }
}

extern "C" void kernel_launch(void* const* d_in, const int* in_sizes, int n_in,
                              void* d_out, int out_size, void* d_ws, size_t ws_size,
                              hipStream_t stream) {
  const float* h1 = (const float*)d_in[0];
  const float* h2 = (const float*)d_in[1];
  const int*   sI = (const int*)d_in[2];
  const int*   dI = (const int*)d_in[3];
  const float* W0 = (const float*)d_in[4];
  const float* b0 = (const float*)d_in[5];
  const float* W1 = (const float*)d_in[6];
  const float* b1 = (const float*)d_in[7];
  const float* W2 = (const float*)d_in[8];
  const float* b2 = (const float*)d_in[9];
  float* out = (float*)d_out;
  _Float16* wsh = (_Float16*)d_ws;
  int E = in_sizes[2];

  prep_w_36971078484654<<<(WS_HALVES + 255) / 256, 256, 0, stream>>>(W0, W1, W2, wsh);

  int nblk = (E + EPB - 1) / EPB;                     // 12,500 for E=1.6M
  edge_mlp_36971078484654<<<nblk, THREADS, 0, stream>>>(
      h1, h2, sI, dI, b0, b1, b2, wsh, out, E);
}
